// MeshedMemoryMultiHeadAttention_new_37563783970980
// MI455X (gfx1250) — matrix operand order repaired and checked by reading, not compiled, not run on hardware
//
#include <hip/hip_runtime.h>
#include <stdint.h>

#define B_   4
#define S_   1024
#define D_   1024
#define H_   16
#define DK_  64
#define NEGV (-1e9f)

typedef __attribute__((ext_vector_type(8)))  _Float16 v8h_t;
typedef __attribute__((ext_vector_type(16))) _Float16 v16h_t;
typedef __attribute__((ext_vector_type(8)))  float    v8f_t;
typedef __attribute__((ext_vector_type(4)))  unsigned v4u_t;
typedef __attribute__((ext_vector_type(8)))  unsigned v8u_t;

// ---------- fragment helpers (wave32 WMMA 16x16x32 f16) ----------
__device__ __forceinline__ v16h_t cat16(v8h_t a, v8h_t b) {
  return __builtin_shufflevector(a, b, 0,1,2,3,4,5,6,7,8,9,10,11,12,13,14,15);
}
// B fragment: lane n<16 -> column n, K=0..15 contiguous; lane>=16 -> K=16..31.
// Caller passes row pointer already offset by 16*hsel.
__device__ __forceinline__ v16h_t ld_bfrag(const _Float16* p) {
  v8h_t lo = *(const v8h_t*)p;
  v8h_t hi = *(const v8h_t*)(p + 8);
  return cat16(lo, hi);
}
// A fragment: lane L -> row L%16; elements 0..15 = K kbase+16*hsel.. contiguous
__device__ __forceinline__ v16h_t ld_afrag(const _Float16* row, int kbase, int hsel) {
  v8h_t lo = *(const v8h_t*)(row + kbase + 16*hsel);
  v8h_t hi = *(const v8h_t*)(row + kbase + 16*hsel + 8);
  return cat16(lo, hi);
}
__device__ __forceinline__ v8f_t wmma16(v16h_t a, v16h_t b, v8f_t c) {
  return __builtin_amdgcn_wmma_f32_16x16x32_f16(false, a, false, b, (short)0, c, false, false);
}
// deterministic per-element bernoulli(0.3) hash (threefry stand-in)
__device__ __forceinline__ bool bern_hit(unsigned idx) {
  unsigned h = idx * 0x9E3779B9u;
  h ^= h >> 16; h *= 0x7feb352du; h ^= h >> 15; h *= 0x846ca68bu; h ^= h >> 16;
  return h < 1288490189u; // floor(0.3 * 2^32)
}

// ---------- Tensor Data Mover: 2D tile (rows x cols f16) global -> LDS ----------
// D# per CDNA5 ISA ch.8: group0 = {count/type/lds_addr/global_addr},
// group1 = {data_size, tensor dims/strides, tile dims}. 2-group form (<=2D tile).
__device__ __forceinline__ void tdm_load_tile_f16(const _Float16* gsrc,
                                                  const _Float16* lds_dst,
                                                  unsigned tile_cols,   // tile_dim0 (elems)
                                                  unsigned tile_rows,   // tile_dim1
                                                  unsigned row_stride)  // tensor_dim0_stride (elems)
{
  unsigned long long ga = (unsigned long long)(uintptr_t)gsrc;
  unsigned ldsoff = (unsigned)(unsigned long long)(uintptr_t)lds_dst; // flat LDS addr low bits = LDS offset
  v4u_t g0;
  g0[0] = 1u;                                   // count=1, user descriptor
  g0[1] = ldsoff;                               // lds_addr
  g0[2] = (unsigned)(ga & 0xffffffffu);         // global_addr[31:0]
  g0[3] = (unsigned)((ga >> 32) & 0x1ffffffu)   // global_addr[56:32]
        | (2u << 30);                           // type = 2 ("image")
  const unsigned td0 = 1u << 20;                // tensor_dim0 (ample, tile in-bounds)
  const unsigned td1 = 1u << 20;                // tensor_dim1
  v8u_t g1;
  g1[0] = (1u << 16);                           // data_size = 1 -> 2 bytes; no pad/iter/multicast
  g1[1] = (td0 & 0xffffu) << 16;                // [31:16] = tensor_dim0[15:0]
  g1[2] = (td0 >> 16) | ((td1 & 0xffffu) << 16);
  g1[3] = (td1 >> 16) | (tile_cols << 16);      // [31:16] = tile_dim0
  g1[4] = tile_rows;                            // tile_dim1 (tile_dim2 = 0)
  g1[5] = row_stride;                           // tensor_dim0_stride[31:0]
  g1[6] = 0;                                    // stride hi + dim1_stride lo
  g1[7] = 0;
  asm volatile("tensor_load_to_lds %0, %1" : : "s"(g0), "s"(g1) : "memory");
}

// ---------- kernel 0: x fp32 -> f16 ----------
__global__ void cvt_x_kernel(const float* __restrict__ x, unsigned short* Xh_u, int n) {
  int i = blockIdx.x * blockDim.x + threadIdx.x;
  if (i < n) ((_Float16*)Xh_u)[i] = (_Float16)x[i];
}

// ---------- kernel 1: W [K][N] fp32 -> Wt [N][K] f16 (LDS-tiled transpose) ----------
__global__ void transpose_cvt_kernel(const float* __restrict__ Wq, const float* __restrict__ Wk,
                                     const float* __restrict__ Wv, const float* __restrict__ Wo,
                                     unsigned short* Wqt_u, unsigned short* Wkt_u,
                                     unsigned short* Wvt_u, unsigned short* Wot_u) {
  __shared__ float tile[32][33];
  int z = blockIdx.z;
  const float* src = (z == 0) ? Wq : (z == 1) ? Wk : (z == 2) ? Wv : Wo;
  _Float16* dst = (_Float16*)((z == 0) ? Wqt_u : (z == 1) ? Wkt_u : (z == 2) ? Wvt_u : Wot_u);
  int x0 = blockIdx.x * 32;  // N tile
  int y0 = blockIdx.y * 32;  // K tile
  for (int r = threadIdx.y; r < 32; r += 8)
    tile[r][threadIdx.x] = src[(size_t)(y0 + r) * D_ + x0 + threadIdx.x];
  __syncthreads();
  for (int r = threadIdx.y; r < 32; r += 8)
    dst[(size_t)(x0 + r) * D_ + y0 + threadIdx.x] = (_Float16)tile[threadIdx.x][r];
}

// ---------- shared GEMM core: 16x64 per wave, TDM double-buffered weight tiles ----------
// Block = 4 waves, all waves share the 64x32 weight tile (TDM once, consume via LDS).
__device__ __forceinline__ void gemm_core(const _Float16* __restrict__ arow,
                                          const _Float16* __restrict__ Wtile_src, // Wt + nb*D_
                                          _Float16* lds, // [2][64*32]
                                          int wave, int l16, int hsel,
                                          v8f_t c[4]) {
  const int NITER = D_ / 32;
  if (wave == 0) {
    tdm_load_tile_f16(Wtile_src, lds, 32, 64, D_);
    __builtin_amdgcn_s_wait_tensorcnt(0);
  }
  __syncthreads();
  for (int i = 0; i < NITER; ++i) {
    int kk = i * 32;
    if (wave == 0 && i + 1 < NITER)
      tdm_load_tile_f16(Wtile_src + kk + 32, lds + ((i + 1) & 1) * (64 * 32), 32, 64, D_);

    v16h_t a = ld_afrag(arow, kk, hsel);
    const _Float16* wl = lds + (i & 1) * (64 * 32);
    v16h_t w0 = ld_bfrag(wl + (l16) * 32 + 16 * hsel);
    v16h_t w1 = ld_bfrag(wl + (16 + l16) * 32 + 16 * hsel);
    v16h_t w2 = ld_bfrag(wl + (32 + l16) * 32 + 16 * hsel);
    v16h_t w3 = ld_bfrag(wl + (48 + l16) * 32 + 16 * hsel);
    c[0] = wmma16(a, w0, c[0]);
    c[1] = wmma16(a, w1, c[1]);
    c[2] = wmma16(a, w2, c[2]);
    c[3] = wmma16(a, w3, c[3]);
    if (kk + 32 < D_) __builtin_prefetch((const void*)(arow + kk + 32), 0, 3);

    if (wave == 0 && i + 1 < NITER) __builtin_amdgcn_s_wait_tensorcnt(0);
    __syncthreads();
  }
}

// ---------- kernel 2: Q/K/V = relu(x @ W + b), layout-specialized stores ----------
__global__ __launch_bounds__(128) void gemm_qkv_kernel(
    const unsigned short* Xh_u,
    const unsigned short* Wqt_u, const unsigned short* Wkt_u, const unsigned short* Wvt_u,
    const float* __restrict__ bq, const float* __restrict__ bk, const float* __restrict__ bv,
    unsigned short* Qh_u, unsigned short* Kh_u, unsigned short* Vt_u) {
  __shared__ _Float16 wtile[2][64 * 32];
  int which = blockIdx.z;
  const _Float16* Xh = (const _Float16*)Xh_u;
  const _Float16* Wt = (const _Float16*)((which == 0) ? Wqt_u : (which == 1) ? Wkt_u : Wvt_u);
  const float* bias  = (which == 0) ? bq : (which == 1) ? bk : bv;
  _Float16* out      = (_Float16*)((which == 0) ? Qh_u : (which == 1) ? Kh_u : Vt_u);

  int lane = threadIdx.x & 31, wave = threadIdx.x >> 5;
  int l16 = lane & 15, hsel = lane >> 4;
  int m0 = blockIdx.y * 64 + wave * 16;   // token rows
  int nb = blockIdx.x * 64;               // output features
  const _Float16* arow = Xh + (size_t)(m0 + l16) * D_;

  v8f_t c[4] = {{0}, {0}, {0}, {0}};
  gemm_core(arow, Wt + (size_t)nb * D_, &wtile[0][0], wave, l16, hsel, c);

  int b = m0 >> 10;          // m0 / S_
  int srow = m0 & (S_ - 1);  // token within sequence
  if (which < 2) {  // Q,K: [b][h][s][d] row-major per head
#pragma unroll
    for (int t = 0; t < 4; ++t) {
      int n = nb + 16 * t + l16;
      int hh = n >> 6, d = n & 63;
      float bsc = bias[n];
      _Float16* op = out + (size_t)(b * H_ + hh) * S_ * DK_ + d;
#pragma unroll
      for (int i = 0; i < 8; ++i) {
        int s = srow + 8 * hsel + i;
        op[(size_t)s * DK_] = (_Float16)fmaxf(c[t][i] + bsc, 0.f);
      }
    }
  } else {  // V stored transposed: [b][h][d][s] -> contiguous 16B stores
#pragma unroll
    for (int t = 0; t < 4; ++t) {
      int n = nb + 16 * t + l16;
      int hh = n >> 6, d = n & 63;
      float bsc = bias[n];
      v8h_t pk;
#pragma unroll
      for (int i = 0; i < 8; ++i) pk[i] = (_Float16)fmaxf(c[t][i] + bsc, 0.f);
      size_t off = ((size_t)(b * H_ + hh) * DK_ + d) * S_ + srow + 8 * hsel;
      *(v8h_t*)(out + off) = pk;
    }
  }
}

// ---------- kernel 3: flash attention, S^T = K·Q^T formulation ----------
__global__ __launch_bounds__(128) void attn_kernel(
    const unsigned short* Qh_u, const unsigned short* Kh_u, const unsigned short* Vt_u,
    const float* __restrict__ pos, const unsigned char* __restrict__ maskb,
    const int* __restrict__ decPtr, unsigned short* Ctx_u) {
  const _Float16* Qh = (const _Float16*)Qh_u;
  const _Float16* Kh = (const _Float16*)Kh_u;
  const _Float16* Vt = (const _Float16*)Vt_u;

  int lane = threadIdx.x & 31, wave = threadIdx.x >> 5;
  int l16 = lane & 15, hsel = lane >> 4;
  int b = blockIdx.z, h = blockIdx.y;
  int q0 = (blockIdx.x * 4 + wave) * 16;
  int q = q0 + l16;

  size_t head = (size_t)(b * H_ + h) * S_ * DK_;
  const _Float16* Kbase = Kh + head;
  const _Float16* Vbase = Vt + (size_t)(b * H_ + h) * DK_ * S_;

  // Q^T B-fragments (d 0..31 and 32..63), fully contiguous per lane
  const _Float16* Qr = Qh + head + (size_t)q * DK_;
  v16h_t qb0 = ld_bfrag(Qr + 16 * hsel);
  v16h_t qb1 = ld_bfrag(Qr + 32 + 16 * hsel);

  v8f_t o[4] = {{0}, {0}, {0}, {0}};  // O^T tiles, d rows 16t..16t+15, q cols
  float mrun = -1e30f, lrun = 0.f;
  int dec = decPtr[0];
  int jmax = dec ? (q0 + 16) : S_;

  const float* posr = pos + ((size_t)b * S_ + q) * S_;
  const unsigned char* mrow = maskb + (size_t)b * S_;
  unsigned idxbase = ((unsigned)(b * H_ + h) * (unsigned)S_ + (unsigned)q) * (unsigned)S_;

  for (int j0 = 0; j0 < jmax; j0 += 32) {
    const _Float16* kr0 = Kbase + (size_t)(j0 + l16) * DK_;
    const _Float16* kr1 = Kbase + (size_t)(j0 + 16 + l16) * DK_;
    v8f_t s0 = {0};
    s0 = wmma16(ld_afrag(kr0, 0, hsel),  qb0, s0);
    s0 = wmma16(ld_afrag(kr0, 32, hsel), qb1, s0);
    v8f_t s1 = {0};
    s1 = wmma16(ld_afrag(kr1, 0, hsel),  qb0, s1);
    s1 = wmma16(ld_afrag(kr1, 32, hsel), qb1, s1);

    int k0 = j0 + 8 * hsel;        // keys for s0 elements i=0..7
    int k1 = j0 + 16 + 8 * hsel;   // keys for s1
    float4 pA = *(const float4*)(posr + k0);
    float4 pB = *(const float4*)(posr + k0 + 4);
    float4 pC = *(const float4*)(posr + k1);
    float4 pD = *(const float4*)(posr + k1 + 4);
    float pa[8] = {pA.x, pA.y, pA.z, pA.w, pB.x, pB.y, pB.z, pB.w};
    float pb[8] = {pC.x, pC.y, pC.z, pC.w, pD.x, pD.y, pD.z, pD.w};
    uint64_t mA = *(const uint64_t*)(mrow + k0);
    uint64_t mB = *(const uint64_t*)(mrow + k1);

    float f0[8], f1[8];
#pragma unroll
    for (int i = 0; i < 8; ++i) {
      int keyA = k0 + i, keyB = k1 + i;
      float vA = s0[i] * 0.125f + pa[i];
      float vB = s1[i] * 0.125f + pb[i];
      if (bern_hit(idxbase + (unsigned)keyA)) vA += NEGV;
      if (bern_hit(idxbase + (unsigned)keyB)) vB += NEGV;
      if ((mA >> (8 * i)) & 0xffull) vA = NEGV;
      if ((mB >> (8 * i)) & 0xffull) vB = NEGV;
      if (dec && keyA > q) vA = NEGV;
      if (dec && keyB > q) vB = NEGV;
      f0[i] = vA; f1[i] = vB;
    }
    // per-query (= per lane-pair) online softmax
    float bm = -1e30f;
#pragma unroll
    for (int i = 0; i < 8; ++i) bm = fmaxf(bm, fmaxf(f0[i], f1[i]));
    bm = fmaxf(bm, __shfl_xor(bm, 16, 32));
    float mn = fmaxf(mrun, bm);
    float corr = __expf(mrun - mn);
    mrun = mn;
    float rs = 0.f;
#pragma unroll
    for (int i = 0; i < 8; ++i) {
      f0[i] = __expf(f0[i] - mn);
      f1[i] = __expf(f1[i] - mn);
      rs += f0[i] + f1[i];
    }
    rs += __shfl_xor(rs, 16, 32);
    lrun = lrun * corr + rs;
#pragma unroll
    for (int t = 0; t < 4; ++t)
#pragma unroll
      for (int i = 0; i < 8; ++i) o[t][i] *= corr;

    // assemble P^T B-fragment (K=32 keys) with lane-pair shuffles only
    v16h_t pf;
#pragma unroll
    for (int i = 0; i < 8; ++i) {
      float x0 = __shfl_xor(f0[i], 16, 32);
      float x1 = __shfl_xor(f1[i], 16, 32);
      pf[i]     = (_Float16)(hsel ? x1 : f0[i]);   // K=e:   keys j0+e   / j0+16+e
      pf[i + 8] = (_Float16)(hsel ? f1[i] : x0);   // K=8+e: keys j0+8+e / j0+24+e
    }
    // O^T += V^T(16d x 32k) · P^T(32k x 16q)
#pragma unroll
    for (int t = 0; t < 4; ++t) {
      const _Float16* vr = Vbase + (size_t)(16 * t + l16) * S_;
      v16h_t vf = ld_afrag(vr, j0, hsel);
      o[t] = wmma16(vf, pf, o[t]);
    }
  }

  float inv = 1.0f / lrun;
  _Float16* Ctx = (_Float16*)Ctx_u;
#pragma unroll
  for (int t = 0; t < 4; ++t) {
    v8h_t pk;
#pragma unroll
    for (int i = 0; i < 8; ++i) pk[i] = (_Float16)(o[t][i] * inv);
    size_t off = ((size_t)b * S_ + q) * (H_ * DK_) + h * DK_ + 16 * t + 8 * hsel;
    *(v8h_t*)(Ctx + off) = pk;
  }
}

// ---------- kernel 4: out = relu(ctx @ Wo + bo), fp32 output ----------
__global__ __launch_bounds__(128) void gemm_out_kernel(
    const unsigned short* Ctx_u, const unsigned short* Wot_u,
    const float* __restrict__ bo, float* __restrict__ out) {
  __shared__ _Float16 wtile[2][64 * 32];
  const _Float16* Ctx = (const _Float16*)Ctx_u;
  const _Float16* Wt  = (const _Float16*)Wot_u;
  int lane = threadIdx.x & 31, wave = threadIdx.x >> 5;
  int l16 = lane & 15, hsel = lane >> 4;
  int m0 = blockIdx.y * 64 + wave * 16;
  int nb = blockIdx.x * 64;
  const _Float16* arow = Ctx + (size_t)(m0 + l16) * D_;

  v8f_t c[4] = {{0}, {0}, {0}, {0}};
  gemm_core(arow, Wt + (size_t)nb * D_, &wtile[0][0], wave, l16, hsel, c);

#pragma unroll
  for (int t = 0; t < 4; ++t) {
    int n = nb + 16 * t + l16;
    float bsc = bo[n];
#pragma unroll
    for (int i = 0; i < 8; ++i) {
      int m = m0 + 8 * hsel + i;
      out[(size_t)m * D_ + n] = fmaxf(c[t][i] + bsc, 0.f);
    }
  }
}

extern "C" void kernel_launch(void* const* d_in, const int* in_sizes, int n_in,
                              void* d_out, int out_size, void* d_ws, size_t ws_size,
                              hipStream_t stream) {
  const float* x            = (const float*)d_in[0];
  const unsigned char* mskb = (const unsigned char*)d_in[1];
  const float* pos          = (const float*)d_in[2];
  const int* dec            = (const int*)d_in[3];
  const float* Wq = (const float*)d_in[4];  const float* bq = (const float*)d_in[5];
  const float* Wk = (const float*)d_in[6];  const float* bk = (const float*)d_in[7];
  const float* Wv = (const float*)d_in[8];  const float* bv = (const float*)d_in[9];
  const float* Wo = (const float*)d_in[10]; const float* bo = (const float*)d_in[11];
  float* out = (float*)d_out;

  const size_t nTok = (size_t)B_ * S_;             // 4096
  const size_t nQKV = (size_t)B_ * H_ * S_ * DK_;  // 4194304
  unsigned short* ws  = (unsigned short*)d_ws;
  unsigned short* Xh  = ws;
  unsigned short* Wqt = Xh  + nTok * D_;
  unsigned short* Wkt = Wqt + (size_t)D_ * D_;
  unsigned short* Wvt = Wkt + (size_t)D_ * D_;
  unsigned short* Wot = Wvt + (size_t)D_ * D_;
  unsigned short* Qh  = Wot + (size_t)D_ * D_;
  unsigned short* Kh  = Qh  + nQKV;
  unsigned short* Vt  = Kh  + nQKV;
  unsigned short* Ctx = Vt  + nQKV;
  size_t needed = (size_t)(Ctx + nTok * D_ - ws) * sizeof(unsigned short);
  if (ws_size < needed) return;

  int nx = (int)(nTok * D_);
  cvt_x_kernel<<<dim3((nx + 255) / 256), 256, 0, stream>>>(x, Xh, nx);
  transpose_cvt_kernel<<<dim3(32, 32, 4), dim3(32, 8), 0, stream>>>(
      Wq, Wk, Wv, Wo, Wqt, Wkt, Wvt, Wot);
  gemm_qkv_kernel<<<dim3(16, 64, 3), 128, 0, stream>>>(
      Xh, Wqt, Wkt, Wvt, bq, bk, bv, Qh, Kh, Vt);
  attn_kernel<<<dim3(S_ / 64, H_, B_), 128, 0, stream>>>(
      Qh, Kh, Vt, pos, mskb, dec, Ctx);
  gemm_out_kernel<<<dim3(16, 64), 128, 0, stream>>>(Ctx, Wot, bo, out);
}
